// SpectralPatchRVQ_19043884990999
// MI455X (gfx1250) — matrix operand order repaired and checked by reading, not compiled, not run on hardware
//
#include <hip/hip_runtime.h>

typedef __attribute__((ext_vector_type(16))) __bf16 v16bf;
typedef __attribute__((ext_vector_type(8)))  float  v8f;
typedef __attribute__((ext_vector_type(4)))  unsigned int v4u;
typedef __attribute__((ext_vector_type(8)))  int v8i_;
typedef __attribute__((ext_vector_type(4)))  int v4i_;

#define RR 4
#define KK 1024
#define DD 32
#define NTOK 32768   // 64 * 512 tokens

// ---- wave32 XOR shuffles via ds_swizzle (compile-time mask, branch-free) ----
template <int XM>
__device__ __forceinline__ int swzi(int x) {
  return __builtin_amdgcn_ds_swizzle(x, (XM << 10) | 31);   // xor_mask=XM, and_mask=31
}
template <int XM>
__device__ __forceinline__ float swzf(float x) {
  return __int_as_float(swzi<XM>(__float_as_int(x)));
}

// ---------------------------------------------------------------------------
// Pack codebook to bf16 (code-major, 32 contiguous dims) + per-code sq norms.
// ---------------------------------------------------------------------------
__global__ __launch_bounds__(256) void rvq_pack(const float* __restrict__ embed,
                                                __bf16* __restrict__ cb,
                                                float* __restrict__ cnorm) {
  int c = blockIdx.x * blockDim.x + threadIdx.x;   // 0 .. R*K-1
  if (c >= RR * KK) return;
  const float* e = embed + (size_t)c * DD;
  float s = 0.f;
#pragma unroll
  for (int d = 0; d < DD; ++d) {
    float f = e[d];
    s += f * f;
    cb[(size_t)c * DD + d] = (__bf16)f;
  }
  cnorm[c] = s;
}

// ---------------------------------------------------------------------------
// TDM: DMA one 64KB codebook slab (16384 dwords) global -> LDS with a single
// tensor_load_to_lds descriptor. Tracked on TENSORcnt. Issued by one wave.
// (6-arg builtin variant: g0, g1, g2, g3, g4, cpol)
// ---------------------------------------------------------------------------
__device__ __forceinline__ void tdm_load_slab(const void* gsrc, void* slab) {
  unsigned long long ga = (unsigned long long)gsrc;
  unsigned lds_off = (unsigned)(uintptr_t)slab;          // low 32 bits = LDS offset
  v4u g0;
  g0[0] = 1u;                                            // count=1 (valid), user mode
  g0[1] = lds_off;                                       // lds_addr
  g0[2] = (unsigned)(ga & 0xFFFFFFFFu);                  // global_addr[31:0]
  g0[3] = (unsigned)((ga >> 32) & 0x01FFFFFFu)           // global_addr[56:32]
        | (2u << 30);                                    // type = 2 ("image")
  v8i_ g1;
  g1[0] = (int)(2u << 16);       // workgroup_mask=0, data_size=2 (4 bytes)
  g1[1] = (int)(16384u << 16);   // tensor_dim0[15:0]=16384 (bits 63:48)
  g1[2] = (int)(1u << 16);       // tensor_dim0[31:16]=0, tensor_dim1[15:0]=1
  g1[3] = (int)(16384u << 16);   // tensor_dim1[31:16]=0, tile_dim0=16384
  g1[4] = 1;                     // tile_dim1=1, tile_dim2=0
  g1[5] = 16384;                 // tensor_dim0_stride[31:0]=16384
  g1[6] = 0;                     // stride0[47:32]=0, stride1[15:0]=0
  g1[7] = 0;
  v4i_ z4 = {0, 0, 0, 0};        // dims >2 unused
  v8i_ z8 = {0, 0, 0, 0, 0, 0, 0, 0};
  __builtin_amdgcn_tensor_load_to_lds(g0, g1, z4, z4, z8, 0);
}

// ---------------------------------------------------------------------------
// Main RVQ kernel: one wave32 per 16-token tile; 8 waves / block share LDS
// codebook slabs, TDM double-buffered across the 4 RVQ stages.
// Residual kept in registers in WMMA A-operand layout:
//   lane l<16 : row l,      dims {0..7, 16..23}
//   lane l>=16: row l-16,   dims {8..15, 24..31}
// B-operand: lane n<16: code n dims {0..7, 16..23}; lane n>=16: code n-16 dims {8..15, 24..31}.
// ---------------------------------------------------------------------------
__global__ __launch_bounds__(256) void rvq_main(const float* __restrict__ x,
                                                const float* __restrict__ w,
                                                const float* __restrict__ embed,
                                                const __bf16* __restrict__ cb,
                                                const float* __restrict__ cnorm,
                                                float* __restrict__ out,
                                                float* __restrict__ accg) {
  __shared__ __align__(16) __bf16 sCB[2][KK * DD];     // 2 x 64KB

  const int tid  = threadIdx.x;
  const int lane = tid & 31;
  const int wid  = (blockIdx.x * blockDim.x + tid) >> 5;  // 0..2047
  const int row  = lane & 15;
  const bool lowK = lane < 16;
  const int d0 = lowK ? 0 : 8;
  const int tok = wid * 16 + row;
  const bool tdm_wave = tid < 32;                      // wave 0 drives the TDM

  // kick off stage-0 slab DMA before doing anything else
  if (tdm_wave) tdm_load_slab(cb, &sCB[0][0]);

  // ---- load this lane's 16 dims of its token (two contiguous runs of 8) ----
  const float* xt = x + (size_t)tok * DD + d0;
  float t0[16], rs[16];
#pragma unroll
  for (int j = 0; j < 8; ++j) {
    t0[j]     = xt[j];
    t0[8 + j] = xt[16 + j];
    rs[j]     = t0[j];
    rs[8 + j] = t0[8 + j];
  }

  float* codes_out = out + 1 + (size_t)NTOK * DD;

  // ---------------------------- RVQ stages --------------------------------
#pragma unroll
  for (int r = 0; r < RR; ++r) {
    if (tdm_wave) __builtin_amdgcn_s_wait_tensorcnt(0);
    __syncthreads();                                   // slab[r&1] visible to all
    if (r + 1 < RR && tdm_wave)                        // prefetch next stage slab
      tdm_load_slab(cb + (size_t)(r + 1) * KK * DD, &sCB[(r + 1) & 1][0]);
    const __bf16* slab = sCB[r & 1];

    // residual -> bf16 A operand (register order already matches A layout)
    v16bf A;
#pragma unroll
    for (int j = 0; j < 16; ++j) A[j] = (__bf16)rs[j];

    const float* cnr = cnorm + (size_t)r * KK;

    float bestS[8];
    int   bestI[8];
#pragma unroll
    for (int v = 0; v < 8; ++v) { bestS[v] = 3.4e38f; bestI[v] = 0; }

    // 64 code tiles of 16; one bf16 WMMA each (K = D = 32 fits exactly)
    for (int tile = 0; tile < 64; ++tile) {
      const int col = tile * 16 + row;                 // this lane's column
      const __bf16* bp = slab + col * DD + d0;
      v16bf B;
#pragma unroll
      for (int j = 0; j < 8; ++j) { B[j] = bp[j]; B[8 + j] = bp[16 + j]; }
      v8f acc = {};
      acc = __builtin_amdgcn_wmma_f32_16x16x32_bf16(
          false, A, false, B, (short)0, acc, false, false);
      const float cn = cnr[col];
#pragma unroll
      for (int v = 0; v < 8; ++v) {
        float s = __builtin_fmaf(-2.f, acc[v], cn);    // ||r||^2 const/row: drop
        int tk = s < bestS[v];
        bestS[v] = tk ? s : bestS[v];
        bestI[v] = tk ? col : bestI[v];
      }
    }

    // branch-free argmin reduction across each 16-lane half (xor stays in-half)
#pragma unroll
    for (int v = 0; v < 8; ++v) {
      float s = bestS[v];
      int   i = bestI[v];
#define RED_STEP(M)                                                        \
      { float so = swzf<M>(s); int io = swzi<M>(i);                        \
        int tk = (int)(so < s) | ((int)(so == s) & (int)(io < i));         \
        s = tk ? so : s; i = tk ? io : i; }
      RED_STEP(1) RED_STEP(2) RED_STEP(4) RED_STEP(8)
#undef RED_STEP
      bestI[v] = i;
    }

    // compile-time 8->1 mux: pick slot (row & 7)
    const int rb  = row & 7;
    int t01 = (rb & 1) ? bestI[1] : bestI[0];
    int t23 = (rb & 1) ? bestI[3] : bestI[2];
    int t45 = (rb & 1) ? bestI[5] : bestI[4];
    int t67 = (rb & 1) ? bestI[7] : bestI[6];
    int u0  = (rb & 2) ? t23 : t01;
    int u1  = (rb & 2) ? t67 : t45;
    int mine  = (rb & 4) ? u1 : u0;
    // lanes 0-15 reduced rows 0-7; lanes 16-31 reduced rows 8-15.
    int other = swzi<16>(mine);
    int idx   = ((row < 8) == lowK) ? mine : other;

    if (lowK) codes_out[(size_t)tok * RR + r] = (float)idx;

    // gather chosen code (f32) and update residual in-register
    const float* q = embed + ((size_t)r * KK + idx) * DD + d0;
#pragma unroll
    for (int j = 0; j < 8; ++j) {
      rs[j]     -= q[j];
      rs[8 + j] -= q[16 + j];
    }
  }

  // ------------------------ outputs + weighted MSE ------------------------
  float* xq = out + 1 + (size_t)tok * DD + d0;
#pragma unroll
  for (int j = 0; j < 8; ++j) {
    xq[j]      = t0[j]     - rs[j];
    xq[16 + j] = t0[8 + j] - rs[8 + j];
  }

  // weight index for dim d of token t is t*16 + d/2; this lane covers
  // positions {p0..p0+3, p0+8..p0+11}, both channels each.
  const float* wp = w + (size_t)tok * 16 + (d0 >> 1);
  float num = 0.f, den = 0.f;
#pragma unroll
  for (int j = 0; j < 4; ++j) {
    float w0 = wp[j], w1 = wp[8 + j];
    den += w0 + w1;
    num += (rs[2*j]     * rs[2*j]     + rs[2*j + 1]     * rs[2*j + 1])     * w0;
    num += (rs[8 + 2*j] * rs[8 + 2*j] + rs[8 + 2*j + 1] * rs[8 + 2*j + 1]) * w1;
  }
#define SUM_STEP(M) { num += swzf<M>(num); den += swzf<M>(den); }
  SUM_STEP(1) SUM_STEP(2) SUM_STEP(4) SUM_STEP(8) SUM_STEP(16)
#undef SUM_STEP
  if (lane == 0) {
    atomicAdd(accg + 0, num);
    atomicAdd(accg + 1, den);
  }
}

__global__ void rvq_fin(const float* __restrict__ accg, float* __restrict__ out) {
  out[0] = accg[0] / fmaxf(accg[1], 1e-6f);
}

// ---------------------------------------------------------------------------
extern "C" void kernel_launch(void* const* d_in, const int* in_sizes, int n_in,
                              void* d_out, int out_size, void* d_ws, size_t ws_size,
                              hipStream_t stream) {
  const float* x     = (const float*)d_in[0];   // [64, 8192, 2]
  const float* w     = (const float*)d_in[1];   // [64, 8192]
  const float* embed = (const float*)d_in[2];   // [4, 1024, 32]
  float* out = (float*)d_out;                   // [loss][x_q 1048576][codes 131072]

  char*   ws    = (char*)d_ws;
  __bf16* cb    = (__bf16*)ws;                        // 4*1024*32*2 = 256 KB
  float*  cnorm = (float*)(ws + 256 * 1024);          // 16 KB
  float*  accg  = (float*)(ws + 272 * 1024);          // 8 B (num, den)

  (void)hipMemsetAsync(accg, 0, 2 * sizeof(float), stream);
  rvq_pack<<<dim3(16), dim3(256), 0, stream>>>(embed, cb, cnorm);
  // 2048 tiles of 16 tokens; 8 waves per 256-thread block -> 256 blocks
  rvq_main<<<dim3(256), dim3(256), 0, stream>>>(x, w, embed, cb, cnorm, out, accg);
  rvq_fin<<<dim3(1), dim3(1), 0, stream>>>(accg, out);
}
